// MPNN_LSTM_Attn3_57183194579540
// MI455X (gfx1250) — compile-verified, matrix operand order checked by reading, not run on hardware
//
#include <hip/hip_runtime.h>
#include <hip/hip_bf16.h>
#include <math.h>

// ---------------------------------------------------------------------------
// Problem constants (match reference)
// ---------------------------------------------------------------------------
#define NNODES   10000
#define WINDOW_T 8
#define BATCH    2
#define NFEAT    64
#define NHID     128
#define NTOT     (BATCH * WINDOW_T * NNODES)   // 160000
#define BNROWS   (BATCH * NNODES)              // 20000
#define NEDGE    1000000
#define EPSV     1e-5f

typedef __bf16 bf16;
typedef __attribute__((ext_vector_type(16))) __bf16 v16bf;
typedef __attribute__((ext_vector_type(8)))  __bf16 v8bf;
typedef __attribute__((ext_vector_type(8)))  float  v8f;

// ---------------------------------------------------------------------------
// WMMA bf16 GEMM:  C[M,Nc] = act( A[M,K] @ op(B) + bias )
//   TRANSB==1: B is [Nc,K] row-major (computes A @ B^T, torch Linear)
//   TRANSB==0: B is [K,Nc] row-major
//   ACT: 0 none, 1 relu, 2 tanh
// Block: 256 threads = 8 waves arranged 4(M) x 2(N).
// Each wave: 32x32 output tile = 2x2 WMMA accumulators -> per K-step:
//   8x ds_load_b128 + 4x v_wmma_f32_16x16x32_bf16 (A frags reused over N).
// Block tile: 128(M) x 64(N), K-step 32. LDS ~15 KB of bf16 (pad 40/row).
// ---------------------------------------------------------------------------
template<int TRANSB, int ACT>
__global__ __launch_bounds__(256)
void gemm_bf16_wmma(const float* __restrict__ A, const float* __restrict__ B,
                    const float* __restrict__ bias, float* __restrict__ C,
                    int M, int Nc, int K) {
  __shared__ bf16 As[128 * 40];  // rows m0..m0+127, k-stride 40 (80B, 16B-aligned)
  __shared__ bf16 Bs[64 * 40];   // Bs[n][k] : K-contiguous per output column

  const int tid  = threadIdx.x;
  const int m0   = blockIdx.y * 128;
  const int n0   = blockIdx.x * 64;
  const int w    = tid >> 5;          // wave 0..7
  const int lane = tid & 31;
  const int hh   = lane >> 4;         // half-wave 0/1
  const int l16  = lane & 15;
  const int wm   = (w >> 1) * 32;     // 0,32,64,96
  const int wn   = (w & 1) * 32;      // 0,32

  v8f acc00 = {}, acc01 = {}, acc10 = {}, acc11 = {};

  for (int k0 = 0; k0 < K; k0 += 32) {
    // ---- stage A tile (128 rows x 32 k), f32 -> bf16; 16 elems/thread ----
    {
      int r  = tid >> 1;               // 0..127
      int ks = (tid & 1) * 16;         // 0 or 16
      int gr = m0 + r; if (gr >= M) gr = M - 1;
      const float* src = A + (size_t)gr * K + k0 + ks;
      bf16* dst = &As[r * 40 + ks];
#pragma unroll
      for (int j = 0; j < 16; ++j) dst[j] = (bf16)src[j];
    }
    // ---- stage B tile as Bs[n][k] (64 cols x 32 k); 8 elems/thread ----
    if (TRANSB) {
      int n  = tid >> 2;               // 0..63
      int ks = (tid & 3) * 8;          // 0,8,16,24
      int gn = n0 + n; if (gn >= Nc) gn = Nc - 1;
      const float* src = B + (size_t)gn * K + k0 + ks;
      bf16* dst = &Bs[n * 40 + ks];
#pragma unroll
      for (int j = 0; j < 8; ++j) dst[j] = (bf16)src[j];
    } else {
      int kk = tid >> 3;               // 0..31
      int ns = (tid & 7) * 8;          // 0..56
      const float* src = B + (size_t)(k0 + kk) * Nc + n0 + ns;
#pragma unroll
      for (int j = 0; j < 8; ++j) Bs[(ns + j) * 40 + kk] = (bf16)src[j];
    }
    __syncthreads();

    // ---- fragments (ISA 16-bit layout: lanes0-15 K={i,i+16}, lanes16-31 +8) ----
    const bf16* ar0 = &As[(wm + l16) * 40 + hh * 8];
    const bf16* ar1 = &As[(wm + 16 + l16) * 40 + hh * 8];
    const bf16* br0 = &Bs[(wn + l16) * 40 + hh * 8];
    const bf16* br1 = &Bs[(wn + 16 + l16) * 40 + hh * 8];
    v8bf p, q;
    p = *(const v8bf*)ar0; q = *(const v8bf*)(ar0 + 16);
    v16bf a0 = __builtin_shufflevector(p, q, 0,1,2,3,4,5,6,7,8,9,10,11,12,13,14,15);
    p = *(const v8bf*)ar1; q = *(const v8bf*)(ar1 + 16);
    v16bf a1 = __builtin_shufflevector(p, q, 0,1,2,3,4,5,6,7,8,9,10,11,12,13,14,15);
    p = *(const v8bf*)br0; q = *(const v8bf*)(br0 + 16);
    v16bf b0 = __builtin_shufflevector(p, q, 0,1,2,3,4,5,6,7,8,9,10,11,12,13,14,15);
    p = *(const v8bf*)br1; q = *(const v8bf*)(br1 + 16);
    v16bf b1 = __builtin_shufflevector(p, q, 0,1,2,3,4,5,6,7,8,9,10,11,12,13,14,15);

    acc00 = __builtin_amdgcn_wmma_f32_16x16x32_bf16(false, a0, false, b0, (short)0, acc00, false, false);
    acc01 = __builtin_amdgcn_wmma_f32_16x16x32_bf16(false, a0, false, b1, (short)0, acc01, false, false);
    acc10 = __builtin_amdgcn_wmma_f32_16x16x32_bf16(false, a1, false, b0, (short)0, acc10, false, false);
    acc11 = __builtin_amdgcn_wmma_f32_16x16x32_bf16(false, a1, false, b1, (short)0, acc11, false, false);
    __syncthreads();
  }

  // ---- epilogue: C/D layout VGPR r -> row (tile + r + 8*hh) ----
#pragma unroll
  for (int mi = 0; mi < 2; ++mi) {
#pragma unroll
    for (int ni = 0; ni < 2; ++ni) {
      const v8f* ap = (mi == 0) ? (ni == 0 ? &acc00 : &acc01)
                                : (ni == 0 ? &acc10 : &acc11);
      int col = n0 + wn + ni * 16 + l16;
      if (col < Nc) {
        float bb = bias ? bias[col] : 0.0f;
#pragma unroll
        for (int r = 0; r < 8; ++r) {
          int row = m0 + wm + mi * 16 + r + hh * 8;
          if (row < M) {
            float v = (*ap)[r] + bb;
            if (ACT == 1) v = v > 0.0f ? v : 0.0f;
            if (ACT == 2) v = tanhf(v);
            C[(size_t)row * Nc + col] = v;
          }
        }
      }
    }
  }
}

// ---------------------------------------------------------------------------
// Elementwise / graph kernels
// ---------------------------------------------------------------------------
__global__ void k_fill(float* __restrict__ p, float v, int n) {
  int i = blockIdx.x * 256 + threadIdx.x;
  if (i < n) p[i] = v;
}

__global__ void k_deg(const int* __restrict__ src, const int* __restrict__ dst,
                      const float* __restrict__ wgt, float* __restrict__ deg, int E) {
  int e = blockIdx.x * 256 + threadIdx.x;
  if (e < E) atomicAdd(&deg[dst[e]], wgt[e]);
  (void)src;
}

__global__ void k_dinv(float* __restrict__ deg, int n) {
  int i = blockIdx.x * 256 + threadIdx.x;
  if (i < n) {
    float d = deg[i];
    deg[i] = d > 0.0f ? rsqrtf(fmaxf(d, EPSV)) : 0.0f;
  }
}

// one wave per edge, 4 coalesced floats per lane, atomic scatter-add
__global__ void k_scatter(const int* __restrict__ src, const int* __restrict__ dst,
                          const float* __restrict__ wgt, const float* __restrict__ dinv,
                          const float* __restrict__ m, float* __restrict__ out, int E) {
  int gid = blockIdx.x * 256 + threadIdx.x;
  int e = gid >> 5, lane = gid & 31;
  if (e >= E) return;
  int s = src[e], d = dst[e];
  float norm = dinv[s] * wgt[e] * dinv[d];
  const float* mp = m + (size_t)s * NHID + lane * 4;
  float* op = out + (size_t)d * NHID + lane * 4;
#pragma unroll
  for (int j = 0; j < 4; ++j) atomicAdd(&op[j], mp[j] * norm);
}

__global__ void k_self_bias_relu(const float* __restrict__ m, const float* __restrict__ dinv,
                                 const float* __restrict__ bias, float* __restrict__ out,
                                 int rows) {
  int i = blockIdx.x * 256 + threadIdx.x;
  if (i >= rows * NHID) return;
  int r = i >> 7, c = i & 127;
  float di = dinv[r];
  float v = out[i] + m[i] * di * di + bias[c];
  out[i] = v > 0.0f ? v : 0.0f;
}

__global__ __launch_bounds__(256)
void k_bn_stats(const float* __restrict__ h, float* __restrict__ stats,
                int rows, int chunks) {
  int col = blockIdx.x & 127, chunk = blockIdx.x >> 7, tid = threadIdx.x;
  float s = 0.0f, s2 = 0.0f;
  int stride = chunks * 256;
  for (int r = chunk * 256 + tid; r < rows; r += stride) {
    float v = h[(size_t)r * NHID + col];
    s += v; s2 += v * v;
  }
  __shared__ float sh[256], sh2[256];
  sh[tid] = s; sh2[tid] = s2; __syncthreads();
  for (int off = 128; off > 0; off >>= 1) {
    if (tid < off) { sh[tid] += sh[tid + off]; sh2[tid] += sh2[tid + off]; }
    __syncthreads();
  }
  if (tid == 0) { atomicAdd(&stats[col], sh[0]); atomicAdd(&stats[128 + col], sh2[0]); }
}

__global__ void k_bn_apply(float* __restrict__ h, const float* __restrict__ stats,
                           const float* __restrict__ g, const float* __restrict__ be,
                           int rows) {
  int i = blockIdx.x * 256 + threadIdx.x;
  if (i >= rows * NHID) return;
  int c = i & 127;
  float mean = stats[c] / rows;
  float var  = stats[128 + c] / rows - mean * mean;
  h[i] = (h[i] - mean) * rsqrtf(var + EPSV) * g[c] + be[c];
}

__global__ void k_cat(const float* __restrict__ h1, const float* __restrict__ h2,
                      float* __restrict__ cat, int rows) {
  int i = blockIdx.x * 256 + threadIdx.x;
  if (i >= rows * 2 * NHID) return;
  int r = i >> 8, c = i & 255;
  cat[i] = (c < NHID) ? h1[(size_t)r * NHID + c] : h2[(size_t)r * NHID + c - NHID];
}

// torch LSTM cell (gate order i,f,g,o); xp rows in node order or [t][bn] order
__global__ void k_lstm_cell(const float* __restrict__ xp, const float* __restrict__ hp,
                            const float* __restrict__ bih, const float* __restrict__ bhh,
                            float* __restrict__ hst, float* __restrict__ cst,
                            float* __restrict__ hs_out, int t, int node_order) {
  int i = blockIdx.x * 256 + threadIdx.x;
  if (i >= BNROWS * NHID) return;
  int bn = i >> 7, j = i & 127;
  int xr;
  if (node_order) { int b = bn / NNODES, nj = bn % NNODES; xr = (b * WINDOW_T + t) * NNODES + nj; }
  else            { xr = t * BNROWS + bn; }
  const float* xz = xp + (size_t)xr * (4 * NHID);
  const float* hz = hp + (size_t)bn * (4 * NHID);
  float zi = xz[j]       + hz[j]       + bih[j]       + bhh[j];
  float zf = xz[128 + j] + hz[128 + j] + bih[128 + j] + bhh[128 + j];
  float zg = xz[256 + j] + hz[256 + j] + bih[256 + j] + bhh[256 + j];
  float zo = xz[384 + j] + hz[384 + j] + bih[384 + j] + bhh[384 + j];
  float ig = 1.0f / (1.0f + expf(-zi));
  float fg = 1.0f / (1.0f + expf(-zf));
  float gg = tanhf(zg);
  float og = 1.0f / (1.0f + expf(-zo));
  float c = fg * cst[i] + ig * gg;
  float h = og * tanhf(c);
  cst[i] = c; hst[i] = h;
  hs_out[(size_t)(t * BNROWS + bn) * NHID + j] = h;
}

__global__ void k_attn_score(const float* __restrict__ u, const float* __restrict__ Va,
                             const float* __restrict__ bvp, float* __restrict__ a, int rows) {
  int r = blockIdx.x * 256 + threadIdx.x;
  if (r >= rows) return;
  float s = bvp[0];
  const float* up = u + (size_t)r * 64;
#pragma unroll
  for (int j = 0; j < 64; ++j) s += up[j] * Va[j];
  a[r] = s;
}

__global__ void k_softpool(const float* __restrict__ hs, const float* __restrict__ a,
                           float* __restrict__ pooled) {
  int i = blockIdx.x * 256 + threadIdx.x;
  if (i >= BNROWS * NHID) return;
  int bn = i >> 7, j = i & 127;
  float av[WINDOW_T], mx = -1e30f;
#pragma unroll
  for (int t = 0; t < WINDOW_T; ++t) { av[t] = a[t * BNROWS + bn]; mx = fmaxf(mx, av[t]); }
  float den = 0.0f;
#pragma unroll
  for (int t = 0; t < WINDOW_T; ++t) { av[t] = expf(av[t] - mx); den += av[t]; }
  float s = 0.0f;
#pragma unroll
  for (int t = 0; t < WINDOW_T; ++t)
    s += hs[(size_t)(t * BNROWS + bn) * NHID + j] * av[t];
  pooled[i] = s / den;
}

__global__ void k_zgather(const float* __restrict__ pooled, const float* __restrict__ x,
                          float* __restrict__ zb) {
  int i = blockIdx.x * 256 + threadIdx.x;
  if (i >= BNROWS * (NHID + WINDOW_T * NFEAT)) return;
  int bn = i / 640, c = i % 640;
  if (c < NHID) { zb[i] = pooled[(size_t)bn * NHID + c]; return; }
  int cc = c - NHID, wd = cc >> 6, f = cc & 63;
  int b = bn / NNODES, nj = bn % NNODES;
  int node = (b * WINDOW_T + wd) * NNODES + nj;
  zb[i] = x[(size_t)node * NFEAT + f];
}

__global__ void k_fc2(const float* __restrict__ fc1, const float* __restrict__ Wf2,
                      const float* __restrict__ bf2, float* __restrict__ out) {
  int r = blockIdx.x * 256 + threadIdx.x;
  if (r >= BNROWS) return;
  float s = bf2[0];
  const float* fp = fc1 + (size_t)r * NHID;
#pragma unroll 8
  for (int j = 0; j < NHID; ++j) s += fp[j] * Wf2[j];
  out[r] = s > 0.0f ? s : 0.0f;
}

// ---------------------------------------------------------------------------
// Orchestration
// ---------------------------------------------------------------------------
static inline dim3 gemm_grid(int M, int Nc) { return dim3((Nc + 63) / 64, (M + 127) / 128); }
#define FILL(p, v, n) k_fill<<<((n) + 255) / 256, 256, 0, stream>>>((p), (v), (n))

extern "C" void kernel_launch(void* const* d_in, const int* in_sizes, int n_in,
                              void* d_out, int out_size, void* d_ws, size_t ws_size,
                              hipStream_t stream) {
  const float* x    = (const float*)d_in[0];
  const float* ew   = (const float*)d_in[1];
  const float* W1   = (const float*)d_in[2];
  const float* b1   = (const float*)d_in[3];
  const float* W2   = (const float*)d_in[4];
  const float* b2   = (const float*)d_in[5];
  const float* g1   = (const float*)d_in[6];
  const float* be1  = (const float*)d_in[7];
  const float* g2   = (const float*)d_in[8];
  const float* be2  = (const float*)d_in[9];
  const float* Wih0 = (const float*)d_in[10];
  const float* Whh0 = (const float*)d_in[11];
  const float* bih0 = (const float*)d_in[12];
  const float* bhh0 = (const float*)d_in[13];
  const float* Wih1 = (const float*)d_in[14];
  const float* Whh1 = (const float*)d_in[15];
  const float* bih1 = (const float*)d_in[16];
  const float* bhh1 = (const float*)d_in[17];
  const float* Wa   = (const float*)d_in[18];
  const float* ba   = (const float*)d_in[19];
  const float* Va   = (const float*)d_in[20];
  const float* bv   = (const float*)d_in[21];
  const float* Wf1  = (const float*)d_in[22];
  const float* bf1  = (const float*)d_in[23];
  const float* Wf2  = (const float*)d_in[24];
  const float* bf2  = (const float*)d_in[25];
  const int*   ei   = (const int*)d_in[26];
  const int* esrc = ei;
  const int* edst = ei + NEDGE;

  float* ws = (float*)d_ws;
  size_t o = 0;
  auto alloc = [&](size_t n) { float* p = ws + o; o += n; return p; };
  float* DEG = alloc(NTOT);                    // deg -> dinv in place
  float* M1  = alloc((size_t)NTOT * NHID);     // GCN feature-transform scratch
  float* H1  = alloc((size_t)NTOT * NHID);
  float* H2  = alloc((size_t)NTOT * NHID);
  float* CAT = alloc((size_t)NTOT * 2 * NHID);
  float* XP  = alloc((size_t)NTOT * 4 * NHID); // LSTM input projections (both layers)
  float* HS0 = alloc((size_t)NTOT * NHID);
  float* HS1 = alloc((size_t)NTOT * NHID);
  float* HP  = alloc((size_t)BNROWS * 4 * NHID);
  float* H0S = alloc((size_t)BNROWS * NHID);
  float* C0S = alloc((size_t)BNROWS * NHID);
  float* H1S = alloc((size_t)BNROWS * NHID);
  float* C1S = alloc((size_t)BNROWS * NHID);
  float* U   = alloc((size_t)NTOT * 64);
  float* AAT = alloc(NTOT);
  float* PL  = alloc((size_t)BNROWS * NHID);
  float* ZB  = alloc((size_t)BNROWS * 640);
  float* FC1 = alloc((size_t)BNROWS * NHID);
  float* ST  = alloc(256);
  (void)ws_size; (void)n_in; (void)in_sizes; (void)out_size;

  // ---- degree + symmetric normalization (self-loop weight 1 folded into init) ----
  FILL(DEG, 1.0f, NTOT);
  k_deg<<<(NEDGE + 255) / 256, 256, 0, stream>>>(esrc, edst, ew, DEG, NEDGE);
  k_dinv<<<(NTOT + 255) / 256, 256, 0, stream>>>(DEG, NTOT);

  // ---- GCN layer 1: m = x@W1 ; scatter ; +self +bias ; relu ; BN ----
  gemm_bf16_wmma<0, 0><<<gemm_grid(NTOT, NHID), 256, 0, stream>>>(x, W1, nullptr, M1, NTOT, NHID, NFEAT);
  FILL(H1, 0.0f, NTOT * NHID);
  k_scatter<<<(NEDGE * 32 + 255) / 256, 256, 0, stream>>>(esrc, edst, ew, DEG, M1, H1, NEDGE);
  k_self_bias_relu<<<(NTOT * NHID + 255) / 256, 256, 0, stream>>>(M1, DEG, b1, H1, NTOT);
  FILL(ST, 0.0f, 256);
  k_bn_stats<<<128 * 32, 256, 0, stream>>>(H1, ST, NTOT, 32);
  k_bn_apply<<<(NTOT * NHID + 255) / 256, 256, 0, stream>>>(H1, ST, g1, be1, NTOT);

  // ---- GCN layer 2 ----
  gemm_bf16_wmma<0, 0><<<gemm_grid(NTOT, NHID), 256, 0, stream>>>(H1, W2, nullptr, M1, NTOT, NHID, NHID);
  FILL(H2, 0.0f, NTOT * NHID);
  k_scatter<<<(NEDGE * 32 + 255) / 256, 256, 0, stream>>>(esrc, edst, ew, DEG, M1, H2, NEDGE);
  k_self_bias_relu<<<(NTOT * NHID + 255) / 256, 256, 0, stream>>>(M1, DEG, b2, H2, NTOT);
  FILL(ST, 0.0f, 256);
  k_bn_stats<<<128 * 32, 256, 0, stream>>>(H2, ST, NTOT, 32);
  k_bn_apply<<<(NTOT * NHID + 255) / 256, 256, 0, stream>>>(H2, ST, g2, be2, NTOT);

  // ---- concat [h1, h2] ----
  k_cat<<<(NTOT * 2 * NHID + 255) / 256, 256, 0, stream>>>(H1, H2, CAT, NTOT);

  // ---- LSTM layer 0: batched input projection, then per-step hidden proj + cell ----
  gemm_bf16_wmma<1, 0><<<gemm_grid(NTOT, 4 * NHID), 256, 0, stream>>>(CAT, Wih0, nullptr, XP, NTOT, 4 * NHID, 2 * NHID);
  FILL(H0S, 0.0f, BNROWS * NHID);
  FILL(C0S, 0.0f, BNROWS * NHID);
  for (int t = 0; t < WINDOW_T; ++t) {
    gemm_bf16_wmma<1, 0><<<gemm_grid(BNROWS, 4 * NHID), 256, 0, stream>>>(H0S, Whh0, nullptr, HP, BNROWS, 4 * NHID, NHID);
    k_lstm_cell<<<(BNROWS * NHID + 255) / 256, 256, 0, stream>>>(XP, HP, bih0, bhh0, H0S, C0S, HS0, t, 1);
  }

  // ---- LSTM layer 1 ----
  gemm_bf16_wmma<1, 0><<<gemm_grid(NTOT, 4 * NHID), 256, 0, stream>>>(HS0, Wih1, nullptr, XP, NTOT, 4 * NHID, NHID);
  FILL(H1S, 0.0f, BNROWS * NHID);
  FILL(C1S, 0.0f, BNROWS * NHID);
  for (int t = 0; t < WINDOW_T; ++t) {
    gemm_bf16_wmma<1, 0><<<gemm_grid(BNROWS, 4 * NHID), 256, 0, stream>>>(H1S, Whh1, nullptr, HP, BNROWS, 4 * NHID, NHID);
    k_lstm_cell<<<(BNROWS * NHID + 255) / 256, 256, 0, stream>>>(XP, HP, bih1, bhh1, H1S, C1S, HS1, t, 0);
  }

  // ---- attention over time + pooling ----
  gemm_bf16_wmma<1, 2><<<gemm_grid(NTOT, 64), 256, 0, stream>>>(HS1, Wa, ba, U, NTOT, 64, NHID); // tanh
  k_attn_score<<<(NTOT + 255) / 256, 256, 0, stream>>>(U, Va, bv, AAT, NTOT);
  k_softpool<<<(BNROWS * NHID + 255) / 256, 256, 0, stream>>>(HS1, AAT, PL);

  // ---- concat(pooled, skip) -> FC1(relu) -> FC2(relu) ----
  k_zgather<<<(BNROWS * 640 + 255) / 256, 256, 0, stream>>>(PL, x, ZB);
  gemm_bf16_wmma<1, 1><<<gemm_grid(BNROWS, NHID), 256, 0, stream>>>(ZB, Wf1, bf1, FC1, BNROWS, NHID, 640);
  k_fc2<<<(BNROWS + 255) / 256, 256, 0, stream>>>(FC1, Wf2, bf2, (float*)d_out);
}
